// TransformerBlock_30081950941708
// MI455X (gfx1250) — compile-verified, matrix operand order checked by reading, not run on hardware
//
#include <hip/hip_runtime.h>
#include <hip/hip_bf16.h>
#include <math.h>

// Problem constants (match reference)
#define B_  8
#define C_  512
#define T_  1024
#define NH_ 8
#define HC_ 2048
#define NL_ 6
#define CK_ 64
#define EPS_ 1e-5f

typedef __attribute__((ext_vector_type(16))) __bf16 v16bf;
typedef __attribute__((ext_vector_type(8)))  float  v8f;
typedef unsigned short ushort_t;

union FragBF { v16bf v; uint4 q[2]; unsigned short us[16]; };

// float -> bf16 round-to-nearest-even
__device__ __forceinline__ unsigned short f2bf(float f) {
  unsigned int u = __float_as_uint(f);
  u += 0x7FFFu + ((u >> 16) & 1u);
  return (unsigned short)(u >> 16);
}
__device__ __forceinline__ float bf2f(unsigned short s) {
  return __uint_as_float(((unsigned int)s) << 16);
}

// Load one WMMA 16-bit operand fragment from a K-contiguous LDS row.
// ISA 7.12.2: lane group g = lane>>4 holds K in {8g..8g+7} (elems 0..7)
// and {16+8g..16+8g+7} (elems 8..15). rowPtr must be 16B aligned.
__device__ __forceinline__ v16bf load_frag(const unsigned short* rowPtr, int g) {
  FragBF f;
  f.q[0] = *(const uint4*)(rowPtr + 8 * g);
  f.q[1] = *(const uint4*)(rowPtr + 16 + 8 * g);
  return f.v;
}

// ---------------------------------------------------------------------------
// GEMM (all-bf16 operands, f32 accumulate):
//   acc[b,m,n] = sum_k W[m,k] * X[b,k,n] + bias[m]
//   v = act? gelu(acc) : acc;  v *= out_mask[b,n] (if given)
//   OutF[b,m,n] = v (f32, if given);  OutB[b,m,n] = bf16(v) (if given)
// Tile 128(M) x 128(N), K-step 32, double-buffered LDS, register-pipelined
// global loads. 8 waves: wave wm=0..3 owns M strip 32 (2 m-tiles),
// wn=0..1 owns N strip 64 (4 n-tiles) -> 8 WMMA accumulators per wave.
// ---------------------------------------------------------------------------
#define TM 128
#define TN 128
#define TK 32

__global__ __launch_bounds__(256)
void gemm_wx(const ushort_t* __restrict__ W, const ushort_t* __restrict__ X,
             const float* __restrict__ bias,
             float* __restrict__ OutF, ushort_t* __restrict__ OutB,
             int M, int N, int K, long strideXb, long strideOb,
             const float* __restrict__ out_mask, int act)
{
  __shared__ __align__(16) unsigned short As[2][TM][TK];  // 2 x 8 KB
  __shared__ __align__(16) unsigned short Bs[2][TN][TK];  // 2 x 8 KB

  const int b   = blockIdx.z;
  const int m0  = blockIdx.y * TM;
  const int n0  = blockIdx.x * TN;
  const int tid = threadIdx.x;
  const int wave = tid >> 5, lane = tid & 31;
  const int g = lane >> 4, lm = lane & 15;
  const int wm = wave >> 1, wn = wave & 1;

  const ushort_t* Xb = X + (long)b * strideXb;

  // staging coordinates (constant per thread)
  const int arow = tid >> 1;            // 0..127
  const int acol = (tid & 1) * 16;      // 0 or 16 (2 x uint4 per thread)
  const int bn2  = (tid & 63) * 2;      // n pair 0..126
  const int bkb  = (tid >> 6) * 8;      // k base 0,8,16,24

  uint4 ra0, ra1;
  unsigned int rb[8];

  auto gload = [&](int k0) {
    const ushort_t* asrc = W + (long)(m0 + arow) * K + k0 + acol;
    ra0 = *(const uint4*)asrc;
    ra1 = *(const uint4*)(asrc + 8);
    #pragma unroll
    for (int j = 0; j < 8; ++j)
      rb[j] = *(const unsigned int*)&Xb[(long)(k0 + bkb + j) * N + n0 + bn2];
  };
  auto sstore = [&](int p) {
    *(uint4*)&As[p][arow][acol]     = ra0;
    *(uint4*)&As[p][arow][acol + 8] = ra1;
    #pragma unroll
    for (int j = 0; j < 8; ++j) {
      Bs[p][bn2][bkb + j]     = (unsigned short)rb[j];
      Bs[p][bn2 + 1][bkb + j] = (unsigned short)(rb[j] >> 16);
    }
  };

  v8f acc[2][4] = {};

  gload(0);
  int p = 0;
  for (int k0 = 0; k0 < K; k0 += TK) {
    sstore(p);
    __syncthreads();
    if (k0 + TK < K) gload(k0 + TK);  // overlap next-tile loads with WMMA

    v16bf a0 = load_frag(&As[p][wm * 32 + lm][0], g);
    v16bf a1 = load_frag(&As[p][wm * 32 + 16 + lm][0], g);
    #pragma unroll
    for (int j = 0; j < 4; ++j) {
      v16bf bf = load_frag(&Bs[p][wn * 64 + j * 16 + lm][0], g);
      acc[0][j] = __builtin_amdgcn_wmma_f32_16x16x32_bf16(
          false, a0, false, bf, (short)0, acc[0][j], false, false);
      acc[1][j] = __builtin_amdgcn_wmma_f32_16x16x32_bf16(
          false, a1, false, bf, (short)0, acc[1][j], false, false);
    }
    p ^= 1;
  }

  // epilogue: C layout lane holds n = lm, m = r + 8*g
  #pragma unroll
  for (int j = 0; j < 4; ++j) {
    int n = n0 + wn * 64 + j * 16 + lm;
    float om = out_mask ? out_mask[(long)b * N + n] : 1.0f;
    #pragma unroll
    for (int mi = 0; mi < 2; ++mi) {
      #pragma unroll
      for (int r = 0; r < 8; ++r) {
        int m = m0 + wm * 32 + mi * 16 + r + 8 * g;
        float v = acc[mi][j][r] + bias[m];
        if (act == 1) v = 0.5f * v * (1.0f + erff(v * 0.70710678118654752f));
        v *= om;
        long o = (long)b * strideOb + (long)m * N + n;
        if (OutF) OutF[o] = v;
        if (OutB) OutB[o] = f2bf(v);
      }
    }
  }
}

// ---------------------------------------------------------------------------
// RoPE in-place on bf16 [B, NH*CK, T]: rows i and i+16 (i<16) of each head.
// ---------------------------------------------------------------------------
__global__ void rope_kernel(ushort_t* __restrict__ X) {
  long idx = (long)blockIdx.x * blockDim.x + threadIdx.x;
  const long total = (long)B_ * NH_ * 16 * T_;
  if (idx >= total) return;
  int t = (int)(idx % T_);
  int i = (int)((idx / T_) % 16);
  int h = (int)((idx / ((long)T_ * 16)) % NH_);
  int b = (int)(idx / ((long)T_ * 16 * NH_));
  float theta = powf(10000.0f, -(float)(2 * i) / 32.0f);
  float ang = theta * (float)t;
  float c = cosf(ang), s = sinf(ang);
  long base = ((long)(b * NH_ + h) * CK_) * T_ + t;
  float x0 = bf2f(X[base + (long)i * T_]);
  float x1 = bf2f(X[base + (long)(i + 16) * T_]);
  X[base + (long)i * T_]        = f2bf(x0 * c - x1 * s);
  X[base + (long)(i + 16) * T_] = f2bf(x1 * c + x0 * s);
}

// ---------------------------------------------------------------------------
// Attention pass 1: per (b,h,t) softmax row stats over s of
//   score'(t,s) = mask? q(:,t).k(:,s)/8 : -1e4   (softmax axis = s)
// Block: (t-block of 16) x 8 waves; wave w owns s-chunks w,w+8,... (16 wide).
// Double-buffered K tile, register-pipelined loads, 1 barrier per chunk.
// ---------------------------------------------------------------------------
__global__ __launch_bounds__(256)
void attn_stats(const ushort_t* __restrict__ Q, const ushort_t* __restrict__ Kp,
                const float* __restrict__ mask,
                float* __restrict__ SM, float* __restrict__ SZ)
{
  __shared__ __align__(16) unsigned short Qs[16][CK_];      // 2 KB
  __shared__ __align__(16) unsigned short Ks[2][128][CK_];  // 32 KB
  __shared__ float redM[8][16], redZ[8][16];

  const int t0 = blockIdx.x * 16;
  const int h = blockIdx.y, b = blockIdx.z;
  const int tid = threadIdx.x, wave = tid >> 5, lane = tid & 31;
  const int g = lane >> 4, lm = lane & 15;
  const float scale = 0.125f;  // 1/sqrt(64)

  const ushort_t* q  = Q  + ((long)b * C_ + h * CK_) * T_;
  const ushort_t* k  = Kp + ((long)b * C_ + h * CK_) * T_;
  const float*    mb = mask + (long)b * T_;

  // stage Q tile [16 t][64 c]: dword loads along t, 2-way transpose scatter
  {
    int i2 = (tid & 7) * 2;
    int cb = tid >> 3;             // 0..31
    #pragma unroll
    for (int j = 0; j < 2; ++j) {
      int c = cb + j * 32;
      unsigned int u = *(const unsigned int*)&q[(long)c * T_ + t0 + i2];
      Qs[i2][c]     = (unsigned short)u;
      Qs[i2 + 1][c] = (unsigned short)(u >> 16);
    }
  }

  float rm[8], rz[8], mrow[8];
  #pragma unroll
  for (int r = 0; r < 8; ++r) { rm[r] = -3.0e38f; rz[r] = 0.0f; }
  #pragma unroll
  for (int r = 0; r < 8; ++r) mrow[r] = mb[t0 + r + 8 * g];

  const int s2  = (tid & 63) * 2;
  const int cbk = (tid >> 6) * 16;
  unsigned int rk[16];

  auto kload = [&](int s0) {
    #pragma unroll
    for (int j = 0; j < 16; ++j)
      rk[j] = *(const unsigned int*)&k[(long)(cbk + j) * T_ + s0 + s2];
  };
  auto kstore = [&](int p) {
    #pragma unroll
    for (int j = 0; j < 16; ++j) {
      Ks[p][s2][cbk + j]     = (unsigned short)rk[j];
      Ks[p][s2 + 1][cbk + j] = (unsigned short)(rk[j] >> 16);
    }
  };

  kload(0);
  int p = 0;
  for (int s0 = 0; s0 < T_; s0 += 128) {
    kstore(p);
    __syncthreads();  // also makes Qs visible on first iteration
    if (s0 + 128 < T_) kload(s0 + 128);

    v8f sa = {};
    #pragma unroll
    for (int kk = 0; kk < 2; ++kk) {
      v16bf a  = load_frag(&Qs[lm][kk * 32], g);
      v16bf bf = load_frag(&Ks[p][wave * 16 + lm][kk * 32], g);
      sa = __builtin_amdgcn_wmma_f32_16x16x32_bf16(
          false, a, false, bf, (short)0, sa, false, false);
    }

    float ms = mb[s0 + wave * 16 + lm];
    #pragma unroll
    for (int r = 0; r < 8; ++r) {
      float val = (ms == 0.0f || mrow[r] == 0.0f) ? -10000.0f : sa[r] * scale;
      float cm = val;                       // row max over the 16 s-columns
      for (int o = 1; o < 16; o <<= 1) cm = fmaxf(cm, __shfl_xor(cm, o));
      float nm = fmaxf(rm[r], cm);
      float e = expf(val - nm);
      for (int o = 1; o < 16; o <<= 1) e += __shfl_xor(e, o);
      rz[r] = rz[r] * expf(rm[r] - nm) + e;
      rm[r] = nm;
    }
    p ^= 1;
  }

  if (lm == 0) {  // lanes 0 and 16 hold stats for rows 8g..8g+7
    #pragma unroll
    for (int r = 0; r < 8; ++r) {
      redM[wave][r + 8 * g] = rm[r];
      redZ[wave][r + 8 * g] = rz[r];
    }
  }
  __syncthreads();
  if (tid < 16) {
    float Mx = -3.0e38f;
    for (int w = 0; w < 8; ++w) Mx = fmaxf(Mx, redM[w][tid]);
    float Z = 0.0f;
    for (int w = 0; w < 8; ++w) Z += redZ[w][tid] * expf(redM[w][tid] - Mx);
    long o = ((long)b * NH_ + h) * T_ + t0 + tid;
    SM[o] = Mx; SZ[o] = Z;
  }
}

// ---------------------------------------------------------------------------
// Attention pass 2: ctx[c,s] = sum_t v[c,t] * exp(score'(t,s)-m_t)/Z_t.
// Block owns (b,h,s-block of 64). Loop t in chunks of 32 with double-buffered
// Q/V/P/stat tiles (2 barriers per chunk): phase A computes P (8 waves, one
// 16x16 score tile each) into LDS bf16; phase B does ctx += V * P per wave.
// ---------------------------------------------------------------------------
__global__ __launch_bounds__(256)
void attn_apply(const ushort_t* __restrict__ Q, const ushort_t* __restrict__ Kp,
                const ushort_t* __restrict__ V, const float* __restrict__ SM,
                const float* __restrict__ SZ, const float* __restrict__ mask,
                ushort_t* __restrict__ Ctx)
{
  __shared__ __align__(16) unsigned short Ks[64][CK_];     // 8 KB (once)
  __shared__ __align__(16) unsigned short Qs[2][32][CK_];  // 8 KB
  __shared__ __align__(16) unsigned short Vs[2][CK_][32];  // 8 KB
  __shared__ __align__(16) unsigned short Ps[2][64][32];   // 8 KB
  __shared__ float sm_s[2][32], sz_s[2][32];

  const int s0 = blockIdx.x * 64;
  const int h = blockIdx.y, b = blockIdx.z;
  const int tid = threadIdx.x, wave = tid >> 5, lane = tid & 31;
  const int g = lane >> 4, lm = lane & 15;
  const float scale = 0.125f;

  const ushort_t* q   = Q  + ((long)b * C_ + h * CK_) * T_;
  const ushort_t* k   = Kp + ((long)b * C_ + h * CK_) * T_;
  const ushort_t* v   = V  + ((long)b * C_ + h * CK_) * T_;
  const float*    mb  = mask + (long)b * T_;
  const float*    smp = SM + ((long)b * NH_ + h) * T_;
  const float*    szp = SZ + ((long)b * NH_ + h) * T_;

  // stage K block [64 s][64 c] once: dword loads along s, transpose scatter
  {
    int sl2 = (tid & 31) * 2;
    int cb  = (tid >> 5) * 8;
    #pragma unroll
    for (int j = 0; j < 8; ++j) {
      int c = cb + j;
      unsigned int u = *(const unsigned int*)&k[(long)c * T_ + s0 + sl2];
      Ks[sl2][c]     = (unsigned short)u;
      Ks[sl2 + 1][c] = (unsigned short)(u >> 16);
    }
  }

  const int cw = wave >> 1;  // phase B: c-tile 0..3
  const int sw = wave & 1;   // phase B: s-pair 0..1
  const int th = wave >> 2;  // phase A: t-half 0..1
  const int ss = wave & 3;   // phase A: s-tile 0..3
  const float msS = mb[s0 + ss * 16 + lm];

  const int qi2 = (tid & 15) * 2;
  const int qcb = (tid >> 4) * 4;

  v8f acc[2] = {};
  int p = 0;

  for (int t0 = 0; t0 < T_; t0 += 32) {
    // stage Q [32][64], V [64][32], stats [32] into buffer p
    #pragma unroll
    for (int j = 0; j < 4; ++j) {
      int c = qcb + j;
      unsigned int u = *(const unsigned int*)&q[(long)c * T_ + t0 + qi2];
      Qs[p][qi2][c]     = (unsigned short)u;
      Qs[p][qi2 + 1][c] = (unsigned short)(u >> 16);
      // V tile is [c][t]: t contiguous both sides -> dword copy
      *(unsigned int*)&Vs[p][c][qi2] =
          *(const unsigned int*)&v[(long)c * T_ + t0 + qi2];
    }
    if (tid < 32) { sm_s[p][tid] = smp[t0 + tid]; sz_s[p][tid] = szp[t0 + tid]; }
    __syncthreads();  // also makes Ks visible on first iteration

    // phase A: score tile -> P (bf16) into Ps[p][s][t]
    {
      v8f sa = {};
      #pragma unroll
      for (int kk = 0; kk < 2; ++kk) {
        v16bf a  = load_frag(&Qs[p][th * 16 + lm][kk * 32], g);
        v16bf bf = load_frag(&Ks[ss * 16 + lm][kk * 32], g);
        sa = __builtin_amdgcn_wmma_f32_16x16x32_bf16(
            false, a, false, bf, (short)0, sa, false, false);
      }
      #pragma unroll
      for (int r = 0; r < 8; ++r) {
        int tl = th * 16 + r + 8 * g;
        float mt = mb[t0 + tl];
        float val = (msS == 0.0f || mt == 0.0f) ? -10000.0f : sa[r] * scale;
        float pv = expf(val - sm_s[p][tl]) / sz_s[p][tl];
        Ps[p][ss * 16 + lm][tl] = f2bf(pv);
      }
    }
    __syncthreads();

    // phase B: acc += V * P  (K = 32 over t)
    {
      v16bf a = load_frag(&Vs[p][cw * 16 + lm][0], g);
      #pragma unroll
      for (int j = 0; j < 2; ++j) {
        v16bf bf = load_frag(&Ps[p][(sw * 2 + j) * 16 + lm][0], g);
        acc[j] = __builtin_amdgcn_wmma_f32_16x16x32_bf16(
            false, a, false, bf, (short)0, acc[j], false, false);
      }
    }
    p ^= 1;
  }

  ushort_t* ctx = Ctx + ((long)b * C_ + h * CK_) * T_;
  #pragma unroll
  for (int j = 0; j < 2; ++j) {
    int n = s0 + (sw * 2 + j) * 16 + lm;
    #pragma unroll
    for (int r = 0; r < 8; ++r) {
      int c = cw * 16 + r + 8 * g;
      ctx[(long)c * T_ + n] = f2bf(acc[j][r]);
    }
  }
}

// ---------------------------------------------------------------------------
// LayerNorm over C of (Y + R), fused dual output:
//   OutF (f32, optional mask) and OutB (bf16, optional mask).
// Block: 32 t values; 8 c-groups of 64 channels each; t-coalesced loads.
// ---------------------------------------------------------------------------
__global__ __launch_bounds__(256)
void add_ln(const float* __restrict__ Y, const float* __restrict__ R,
            const float* __restrict__ gamma, const float* __restrict__ beta,
            const float* __restrict__ mask,
            float* __restrict__ OutF, ushort_t* __restrict__ OutB,
            int maskF, int maskB)
{
  __shared__ float s1[8][32], s2[8][32];
  __shared__ float smu[32], srs[32];
  const int t0 = blockIdx.x * 32;
  const int b  = blockIdx.y;
  const int tid = threadIdx.x;
  const int tl = tid & 31;
  const int cg = tid >> 5;  // 0..7
  const long base = (long)b * C_ * T_ + t0 + tl;

  float sum = 0.0f, sq = 0.0f;
  for (int j = 0; j < 64; ++j) {
    int c = cg * 64 + j;
    float vv = Y[base + (long)c * T_] + R[base + (long)c * T_];
    sum += vv; sq += vv * vv;
  }
  s1[cg][tl] = sum; s2[cg][tl] = sq;
  __syncthreads();
  if (cg == 0) {
    float a = 0.0f, qq = 0.0f;
    for (int w = 0; w < 8; ++w) { a += s1[w][tl]; qq += s2[w][tl]; }
    float mu = a / (float)C_;
    float var = qq / (float)C_ - mu * mu;
    smu[tl] = mu;
    srs[tl] = rsqrtf(var + EPS_);
  }
  __syncthreads();
  const float mu = smu[tl], rs = srs[tl];
  const float mk = mask[(long)b * T_ + t0 + tl];
  const float mF = maskF ? mk : 1.0f;
  const float mB = maskB ? mk : 1.0f;
  for (int j = 0; j < 64; ++j) {
    int c = cg * 64 + j;
    long o = base + (long)c * T_;
    float vv = Y[o] + R[o];
    float nv = (vv - mu) * rs * gamma[c] + beta[c];
    if (OutF) OutF[o] = nv * mF;
    if (OutB) OutB[o] = f2bf(nv * mB);
  }
}

// x0 = x * x_mask (broadcast over channels), dual f32 + bf16 output
__global__ void mask_copy(const float* __restrict__ X,
                          const float* __restrict__ mask,
                          float* __restrict__ OutF, ushort_t* __restrict__ OutB)
{
  long i = (long)blockIdx.x * blockDim.x + threadIdx.x;
  const long total = (long)B_ * C_ * T_;
  if (i >= total) return;
  long t = i % T_;
  long b = i / ((long)C_ * T_);
  float v = X[i] * mask[b * T_ + t];
  OutF[i] = v;
  OutB[i] = f2bf(v);
}

// one-shot f32 -> bf16 tensor conversion (weights)
__global__ void cvt_bf16(const float* __restrict__ src, ushort_t* __restrict__ dst,
                         long n)
{
  long i = ((long)blockIdx.x * blockDim.x + threadIdx.x) * 4;
  if (i + 3 >= n) {
    for (long j = i; j < n; ++j) dst[j] = f2bf(src[j]);
    return;
  }
  float4 f = *(const float4*)(src + i);
  dst[i]     = f2bf(f.x);
  dst[i + 1] = f2bf(f.y);
  dst[i + 2] = f2bf(f.z);
  dst[i + 3] = f2bf(f.w);
}

// ---------------------------------------------------------------------------
extern "C" void kernel_launch(void* const* d_in, const int* in_sizes, int n_in,
                              void* d_out, int out_size, void* d_ws, size_t ws_size,
                              hipStream_t stream)
{
  const float* x   = (const float*)d_in[0];
  const float* xm  = (const float*)d_in[1];
  const float* Wq  = (const float*)d_in[2];
  const float* bq  = (const float*)d_in[3];
  const float* Wk  = (const float*)d_in[4];
  const float* bk  = (const float*)d_in[5];
  const float* Wv  = (const float*)d_in[6];
  const float* bv  = (const float*)d_in[7];
  const float* Wo  = (const float*)d_in[8];
  const float* bo  = (const float*)d_in[9];
  const float* W1  = (const float*)d_in[10];
  const float* b1  = (const float*)d_in[11];
  const float* W2  = (const float*)d_in[12];
  const float* b2  = (const float*)d_in[13];
  const float* g1  = (const float*)d_in[14];
  const float* be1 = (const float*)d_in[15];
  const float* g2  = (const float*)d_in[16];
  const float* be2 = (const float*)d_in[17];
  float* out = (float*)d_out;
  (void)in_sizes; (void)n_in; (void)out_size; (void)ws_size;

  const long NX = (long)B_ * C_ * T_;
  const long NH16 = (long)B_ * NH_ * T_;
  const long NWP = (long)NL_ * C_ * C_;    // per projection weight tensor
  const long NWF = (long)NL_ * HC_ * C_;   // per FFN weight tensor

  // f32 region
  float* wsf = (float*)d_ws;
  float* X_  = wsf; wsf += NX;             // residual / current x (f32)
  float* Yb  = wsf; wsf += NX;             // GEMM f32 output for LN
  float* SMb = wsf; wsf += NH16;           // softmax row max
  float* SZb = wsf; wsf += NH16;           // softmax row denom
  // bf16 region
  ushort_t* wsh = (ushort_t*)wsf;
  ushort_t* Xbf = wsh; wsh += NX;          // bf16 copy of x (masked as needed)
  ushort_t* Qb  = wsh; wsh += NX;
  ushort_t* Kb  = wsh; wsh += NX;
  ushort_t* Vb  = wsh; wsh += NX;
  ushort_t* Cx  = wsh; wsh += NX;          // attention context (bf16)
  ushort_t* Hb  = wsh; wsh += (long)B_ * HC_ * T_;  // FFN hidden (bf16, masked)
  ushort_t* Wqb = wsh; wsh += NWP;
  ushort_t* Wkb = wsh; wsh += NWP;
  ushort_t* Wvb = wsh; wsh += NWP;
  ushort_t* Wob = wsh; wsh += NWP;
  ushort_t* W1b = wsh; wsh += NWF;
  ushort_t* W2b = wsh; wsh += NWF;

  const dim3 blk(256);
  const dim3 gP(T_ / TN, C_ / TM, B_);   // C x T projection GEMMs (8,4,8)
  const dim3 gH(T_ / TN, HC_ / TM, B_);  // HC x T GEMM (8,16,8)
  const dim3 gS(T_ / 16, NH_, B_);       // attention stats
  const dim3 gA(T_ / 64, NH_, B_);       // attention apply
  const dim3 gL(T_ / 32, B_);            // layernorm
  const long strideC = (long)C_ * T_, strideH = (long)HC_ * T_;

  // weights -> bf16 (once per launch)
  {
    unsigned gp = (unsigned)((NWP / 4 + 255) / 256);
    unsigned gf = (unsigned)((NWF / 4 + 255) / 256);
    cvt_bf16<<<dim3(gp), blk, 0, stream>>>(Wq, Wqb, NWP);
    cvt_bf16<<<dim3(gp), blk, 0, stream>>>(Wk, Wkb, NWP);
    cvt_bf16<<<dim3(gp), blk, 0, stream>>>(Wv, Wvb, NWP);
    cvt_bf16<<<dim3(gp), blk, 0, stream>>>(Wo, Wob, NWP);
    cvt_bf16<<<dim3(gf), blk, 0, stream>>>(W1, W1b, NWF);
    cvt_bf16<<<dim3(gf), blk, 0, stream>>>(W2, W2b, NWF);
  }

  mask_copy<<<dim3((unsigned)((NX + 255) / 256)), blk, 0, stream>>>(x, xm, X_, Xbf);

  const long nrope = (long)B_ * NH_ * 16 * T_;
  for (int l = 0; l < NL_; ++l) {
    // Q/K/V projections -> bf16
    gemm_wx<<<gP, blk, 0, stream>>>(Wqb + l * (long)C_ * C_, Xbf, bq + l * C_,
                                    nullptr, Qb, C_, T_, C_, strideC, strideC,
                                    nullptr, 0);
    gemm_wx<<<gP, blk, 0, stream>>>(Wkb + l * (long)C_ * C_, Xbf, bk + l * C_,
                                    nullptr, Kb, C_, T_, C_, strideC, strideC,
                                    nullptr, 0);
    gemm_wx<<<gP, blk, 0, stream>>>(Wvb + l * (long)C_ * C_, Xbf, bv + l * C_,
                                    nullptr, Vb, C_, T_, C_, strideC, strideC,
                                    nullptr, 0);

    rope_kernel<<<dim3((unsigned)((nrope + 255) / 256)), blk, 0, stream>>>(Qb);
    rope_kernel<<<dim3((unsigned)((nrope + 255) / 256)), blk, 0, stream>>>(Kb);

    attn_stats<<<gS, blk, 0, stream>>>(Qb, Kb, xm, SMb, SZb);
    attn_apply<<<gA, blk, 0, stream>>>(Qb, Kb, Vb, SMb, SZb, xm, Cx);

    // output projection -> f32, then x = LN(y + x); bf16 copy gets FFN mask
    gemm_wx<<<gP, blk, 0, stream>>>(Wob + l * (long)C_ * C_, Cx, bo + l * C_,
                                    Yb, nullptr, C_, T_, C_, strideC, strideC,
                                    nullptr, 0);
    add_ln<<<gL, blk, 0, stream>>>(Yb, X_, g1 + l * C_, be1 + l * C_, xm,
                                   X_, Xbf, /*maskF=*/0, /*maskB=*/1);

    // FFN: h = gelu(W1 xmask + b1) * mask (bf16); y = (W2 h + b2) * mask (f32)
    gemm_wx<<<gH, blk, 0, stream>>>(W1b + l * (long)HC_ * C_, Xbf, b1 + l * HC_,
                                    nullptr, Hb, HC_, T_, C_, strideC, strideH,
                                    xm, 1);
    gemm_wx<<<gP, blk, 0, stream>>>(W2b + l * (long)HC_ * C_, Hb, b2 + l * C_,
                                    Yb, nullptr, C_, T_, HC_, strideH, strideC,
                                    xm, 0);

    if (l == NL_ - 1)
      add_ln<<<gL, blk, 0, stream>>>(Yb, X_, g2 + l * C_, be2 + l * C_, xm,
                                     out, nullptr, /*maskF=*/1, /*maskB=*/0);
    else
      add_ln<<<gL, blk, 0, stream>>>(Yb, X_, g2 + l * C_, be2 + l * C_, xm,
                                     X_, Xbf, /*maskF=*/0, /*maskB=*/0);
  }
}
